// Perceiver_75625784148257
// MI455X (gfx1250) — compile-verified, matrix operand order checked by reading, not run on hardware
//
#include <hip/hip_runtime.h>

typedef __attribute__((ext_vector_type(16))) _Float16 v16h;
typedef __attribute__((ext_vector_type(8)))  _Float16 h8;
typedef __attribute__((ext_vector_type(8)))  float    v8f;

// CDNA5 16-bit fragment (ISA 7.12.2): for a lane in half h, elements 0..7 hold
// k = 8h..8h+7 and elements 8..15 hold k = 16+8h..16+8h+7.  When the fragment's
// k-axis is contiguous in memory this is exactly two 16-byte loads.
__device__ __forceinline__ v16h load_frag(const _Float16* base, int half) {
  h8 lo = *(const h8*)(base + 8 * half);
  h8 hi = *(const h8*)(base + 16 + 8 * half);
  return __builtin_shufflevector(lo, hi, 0, 1, 2, 3, 4, 5, 6, 7, 8, 9, 10, 11,
                                 12, 13, 14, 15);
}

__device__ __forceinline__ v8f wmma16(v16h a, v16h b, v8f c) {
  return __builtin_amdgcn_wmma_f32_16x16x32_f16(false, a, false, b, (short)0, c,
                                                false, false);
}

// ---------------------------------------------------------------------------
// GEMM: C[b] = act(A[b] @ B[b] + bias + resid).  A row-major [M][K], Bt is B
// TRANSPOSED [N][K] so both fragments are k-contiguous (2x b128 per fragment).
// One wave computes 16x64 (A-fragment reused by 4 WMMAs); 4 waves per block.
// ---------------------------------------------------------------------------
__global__ __launch_bounds__(128) void wmma_gemm(
    const _Float16* __restrict__ A, const _Float16* __restrict__ Bt,
    const float* __restrict__ bias, const float* __restrict__ resid,
    float* __restrict__ Cf, _Float16* __restrict__ Ch, int Mrows, int N, int K,
    long sA, long sB, long sC, int act_gelu, int store_trans) {
  int lane = threadIdx.x & 31, wave = threadIdx.x >> 5;
  int half = lane >> 4, ml = lane & 15;
  int m0 = blockIdx.y * 16;
  int n0 = (blockIdx.x * 4 + wave) * 64;
  const _Float16* Ab = A + (size_t)blockIdx.z * sA;
  const _Float16* Bb = Bt + (size_t)blockIdx.z * sB;
  size_t cOff = (size_t)blockIdx.z * sC;

  const _Float16* arow = Ab + (size_t)(m0 + ml) * K;
  const _Float16* br0 = Bb + (size_t)(n0 + 0 + ml) * K;
  const _Float16* br1 = Bb + (size_t)(n0 + 16 + ml) * K;
  const _Float16* br2 = Bb + (size_t)(n0 + 32 + ml) * K;
  const _Float16* br3 = Bb + (size_t)(n0 + 48 + ml) * K;

  v8f acc0 = {}, acc1 = {}, acc2 = {}, acc3 = {};
  for (int kc = 0; kc < K; kc += 32) {
    v16h a = load_frag(arow + kc, half);
    acc0 = wmma16(a, load_frag(br0 + kc, half), acc0);
    acc1 = wmma16(a, load_frag(br1 + kc, half), acc1);
    acc2 = wmma16(a, load_frag(br2 + kc, half), acc2);
    acc3 = wmma16(a, load_frag(br3 + kc, half), acc3);
  }
#pragma unroll
  for (int nt = 0; nt < 4; ++nt) {
    v8f acc = nt == 0 ? acc0 : nt == 1 ? acc1 : nt == 2 ? acc2 : acc3;
    int n = n0 + nt * 16 + ml;
#pragma unroll
    for (int r = 0; r < 8; ++r) {
      int m = m0 + r + 8 * half;
      float v = acc[r];
      if (bias) v += bias[n];
      if (resid) v += resid[cOff + (size_t)m * N + n];
      if (act_gelu) v = 0.5f * v * (1.0f + erff(v * 0.70710678118654752f));
      if (Cf) Cf[cOff + (size_t)m * N + n] = v;
      if (Ch) {
        if (store_trans)
          Ch[cOff + (size_t)n * Mrows + m] = (_Float16)v;
        else
          Ch[cOff + (size_t)m * N + n] = (_Float16)v;
      }
    }
  }
}

// ---------------------------------------------------------------------------
// Flash attention: wave = 16 query rows x 64 head dim, streams KV in 32s.
// K is row-major [kv][dim] (k-contiguous per lane); V is TRANSPOSED
// [dim][kv] so PV fragments are k-contiguous too.  4 waves per block handle
// 4 consecutive query tiles.  grid = (qrows/64, heads, batch).
// ---------------------------------------------------------------------------
__global__ __launch_bounds__(128) void flash_attn(
    const _Float16* __restrict__ Q, const _Float16* __restrict__ Km,
    const _Float16* __restrict__ Vt, _Float16* __restrict__ O, int kv_len,
    int ldq, int ldk, int ldvt, int ldo, long sQ, long sK, long sVt, long sO,
    float scale) {
  int lane = threadIdx.x & 31, wave = threadIdx.x >> 5;
  int half = lane >> 4, ml = lane & 15;
  int m0 = (blockIdx.x * 4 + wave) * 16;
  int off = blockIdx.y * 64;
  const _Float16* Qb = Q + (size_t)blockIdx.z * sQ;
  const _Float16* Kb = Km + (size_t)blockIdx.z * sK;
  const _Float16* Vb = Vt + (size_t)blockIdx.z * sVt;
  _Float16* Ob = O + (size_t)blockIdx.z * sO;

  const _Float16* qrow = Qb + (size_t)(m0 + ml) * ldq + off;
  v16h qa0 = load_frag(qrow, half);
  v16h qa1 = load_frag(qrow + 32, half);

  // V columns handled by this lane (transposed layout, rows = head-dim cols)
  const _Float16* vr0 = Vb + (size_t)(off + 0 + ml) * ldvt;
  const _Float16* vr1 = Vb + (size_t)(off + 16 + ml) * ldvt;
  const _Float16* vr2 = Vb + (size_t)(off + 32 + ml) * ldvt;
  const _Float16* vr3 = Vb + (size_t)(off + 48 + ml) * ldvt;

  float Mr[8], Lr[8];
  v8f o0 = {}, o1 = {}, o2 = {}, o3 = {};
#pragma unroll
  for (int r = 0; r < 8; ++r) { Mr[r] = -1e30f; Lr[r] = 0.f; }

  __shared__ _Float16 Pst[4][16][40];  // 80B row pitch: 16B-aligned rows

  for (int j0 = 0; j0 < kv_len; j0 += 32) {
    if (j0 + 32 < kv_len) {
      __builtin_prefetch(&Kb[(size_t)(j0 + 32 + ml) * ldk + off], 0, 1);
      __builtin_prefetch(&vr0[j0 + 32], 0, 1);
    }
    // S = Q @ K^T for 32 keys as two 16x16 WMMA tiles
    const _Float16* kr0 = Kb + (size_t)(j0 + ml) * ldk + off;
    const _Float16* kr1 = Kb + (size_t)(j0 + 16 + ml) * ldk + off;
    v8f s0 = {}, s1 = {};
    s0 = wmma16(qa0, load_frag(kr0, half), s0);
    s0 = wmma16(qa1, load_frag(kr0 + 32, half), s0);
    s1 = wmma16(qa0, load_frag(kr1, half), s1);
    s1 = wmma16(qa1, load_frag(kr1 + 32, half), s1);
    // online softmax per query row (row = VGPR index; keys across 16 lanes)
#pragma unroll
    for (int r = 0; r < 8; ++r) {
      float a0 = s0[r] * scale, a1 = s1[r] * scale;
      float mx = fmaxf(a0, a1);
      mx = fmaxf(mx, __shfl_xor(mx, 1, 32));
      mx = fmaxf(mx, __shfl_xor(mx, 2, 32));
      mx = fmaxf(mx, __shfl_xor(mx, 4, 32));
      mx = fmaxf(mx, __shfl_xor(mx, 8, 32));
      float nM = fmaxf(Mr[r], mx);
      float alpha = __expf(Mr[r] - nM);
      float p0 = __expf(a0 - nM), p1 = __expf(a1 - nM);
      float rs = p0 + p1;
      rs += __shfl_xor(rs, 1, 32);
      rs += __shfl_xor(rs, 2, 32);
      rs += __shfl_xor(rs, 4, 32);
      rs += __shfl_xor(rs, 8, 32);
      Lr[r] = Lr[r] * alpha + rs;
      Mr[r] = nM;
      o0[r] *= alpha; o1[r] *= alpha; o2[r] *= alpha; o3[r] *= alpha;
      int m = r + 8 * half;
      Pst[wave][m][ml] = (_Float16)p0;
      Pst[wave][m][16 + ml] = (_Float16)p1;
    }
    __syncthreads();
    v16h pa = load_frag(&Pst[wave][ml][0], half);
    o0 = wmma16(pa, load_frag(vr0 + j0, half), o0);
    o1 = wmma16(pa, load_frag(vr1 + j0, half), o1);
    o2 = wmma16(pa, load_frag(vr2 + j0, half), o2);
    o3 = wmma16(pa, load_frag(vr3 + j0, half), o3);
    __syncthreads();
  }
#pragma unroll
  for (int r = 0; r < 8; ++r) {
    int m = m0 + r + 8 * half;
    float inv = 1.0f / Lr[r];
    Ob[(size_t)m * ldo + off + 0 + ml]  = (_Float16)(o0[r] * inv);
    Ob[(size_t)m * ldo + off + 16 + ml] = (_Float16)(o1[r] * inv);
    Ob[(size_t)m * ldo + off + 32 + ml] = (_Float16)(o2[r] * inv);
    Ob[(size_t)m * ldo + off + 48 + ml] = (_Float16)(o3[r] * inv);
  }
}

// ---------------------------------------------------------------------------
// Fused: fourier pos-enc + concat + token LayerNorm + K/V projection (K=29).
// K written row-major [b][HW][64]; V written TRANSPOSED [b][64][HW].
// ---------------------------------------------------------------------------
#define PH 224
#define PW 224
#define PHW (224 * 224)
#define PB 4
#define IND 29
#define HDI 64

__global__ __launch_bounds__(256) void build_kv(
    const float* __restrict__ data, const float* __restrict__ g,
    const float* __restrict__ bta, const float* __restrict__ wk,
    const float* __restrict__ wv, _Float16* __restrict__ Ko,
    _Float16* __restrict__ Vt) {
  __shared__ float swk[IND * HDI];
  __shared__ float swv[IND * HDI];
  __shared__ float sg[IND], sb[IND];
  for (int i = threadIdx.x; i < IND * HDI; i += 256) {
    swk[i] = wk[i];
    swv[i] = wv[i];
  }
  if (threadIdx.x < IND) {
    sg[threadIdx.x] = g[threadIdx.x];
    sb[threadIdx.x] = bta[threadIdx.x];
  }
  __syncthreads();
  long t = (long)blockIdx.x * 256 + threadIdx.x;
  if (t >= (long)PB * PHW) return;
  int bi = (int)(t / PHW), pix = (int)(t % PHW);
  int h = pix / PW, w = pix % PW;
  float f[IND];
  f[0] = data[((size_t)(bi * 3 + 0) * PH + h) * PW + w];
  f[1] = data[((size_t)(bi * 3 + 1) * PH + h) * PW + w];
  f[2] = data[((size_t)(bi * 3 + 2) * PH + h) * PW + w];
  const float PIF = 3.14159265358979f;
  float cy = -1.f + 2.f * h / (PH - 1);
  float cx = -1.f + 2.f * w / (PW - 1);
#pragma unroll
  for (int n = 0; n < 2; ++n) {
    float c = (n == 0) ? cy : cx;
#pragma unroll
    for (int d = 0; d < 6; ++d) {
      float arg = c * (1.0f + 0.8f * d) * PIF;  // scales = linspace(1,5,6)
      f[3 + n * 13 + d] = __sinf(arg);
      f[3 + n * 13 + 6 + d] = __cosf(arg);
    }
    f[3 + n * 13 + 12] = c;
  }
  float mu = 0.f;
#pragma unroll
  for (int c = 0; c < IND; ++c) mu += f[c];
  mu *= (1.0f / IND);
  float var = 0.f;
#pragma unroll
  for (int c = 0; c < IND; ++c) {
    float d = f[c] - mu;
    var += d * d;
  }
  var *= (1.0f / IND);
  float inv = rsqrtf(var + 1e-5f);
#pragma unroll
  for (int c = 0; c < IND; ++c) f[c] = (f[c] - mu) * inv * sg[c] + sb[c];
  size_t kbase = (size_t)t * HDI;
  size_t vbase = (size_t)bi * HDI * PHW + pix;
  for (int j = 0; j < HDI; ++j) {
    float ak = 0.f, av = 0.f;
#pragma unroll
    for (int c = 0; c < IND; ++c) {
      ak += f[c] * swk[c * HDI + j];
      av += f[c] * swv[c * HDI + j];
    }
    Ko[kbase + j] = (_Float16)ak;
    Vt[vbase + (size_t)j * PHW] = (_Float16)av;
  }
}

// ---------------------------------------------------------------------------
// Q = LN(latents) @ c_wq  (one block per latent row, 64 output cols)
// ---------------------------------------------------------------------------
__global__ __launch_bounds__(64) void q_proj(const float* __restrict__ lat,
                                             const float* __restrict__ g,
                                             const float* __restrict__ bt,
                                             const float* __restrict__ wq,
                                             _Float16* __restrict__ Qo) {
  int row = blockIdx.x, j = threadIdx.x;
  const float* x = lat + (size_t)row * 512;
  __shared__ float red[64];
  float s = 0.f;
  for (int c = j; c < 512; c += 64) s += x[c];
  red[j] = s;
  __syncthreads();
  for (int st = 32; st > 0; st >>= 1) {
    if (j < st) red[j] += red[j + st];
    __syncthreads();
  }
  float mu = red[0] / 512.f;
  __syncthreads();
  s = 0.f;
  for (int c = j; c < 512; c += 64) {
    float d = x[c] - mu;
    s += d * d;
  }
  red[j] = s;
  __syncthreads();
  for (int st = 32; st > 0; st >>= 1) {
    if (j < st) red[j] += red[j + st];
    __syncthreads();
  }
  float inv = rsqrtf(red[0] / 512.f + 1e-5f);
  float acc = 0.f;
  for (int c = 0; c < 512; ++c) {
    float xn = (x[c] - mu) * inv * g[c] + bt[c];
    acc += xn * wq[(size_t)c * 64 + j];
  }
  Qo[(size_t)row * 64 + j] = (_Float16)acc;
}

// ---------------------------------------------------------------------------
// Row LayerNorm (f32 in -> f16 out)
// ---------------------------------------------------------------------------
__global__ __launch_bounds__(256) void ln_rows(const float* __restrict__ X,
                                               const float* __restrict__ g,
                                               const float* __restrict__ bt,
                                               _Float16* __restrict__ Y,
                                               int ncols) {
  int row = blockIdx.x;
  const float* x = X + (size_t)row * ncols;
  __shared__ float red[256];
  float s = 0.f;
  for (int c = threadIdx.x; c < ncols; c += 256) s += x[c];
  red[threadIdx.x] = s;
  __syncthreads();
  for (int st = 128; st > 0; st >>= 1) {
    if (threadIdx.x < st) red[threadIdx.x] += red[threadIdx.x + st];
    __syncthreads();
  }
  float mu = red[0] / ncols;
  __syncthreads();
  s = 0.f;
  for (int c = threadIdx.x; c < ncols; c += 256) {
    float d = x[c] - mu;
    s += d * d;
  }
  red[threadIdx.x] = s;
  __syncthreads();
  for (int st = 128; st > 0; st >>= 1) {
    if (threadIdx.x < st) red[threadIdx.x] += red[threadIdx.x + st];
    __syncthreads();
  }
  float inv = rsqrtf(red[0] / ncols + 1e-5f);
  for (int c = threadIdx.x; c < ncols; c += 256)
    Y[(size_t)row * ncols + c] = (_Float16)((x[c] - mu) * inv * g[c] + bt[c]);
}

// ---------------------------------------------------------------------------
// Head: mean over 512 latents -> LN -> @ h_w (512x2) + h_b
// ---------------------------------------------------------------------------
__global__ __launch_bounds__(512) void head_k(const float* __restrict__ X,
                                              const float* __restrict__ g,
                                              const float* __restrict__ bt,
                                              const float* __restrict__ hw,
                                              const float* __restrict__ hb,
                                              float* __restrict__ out) {
  int bi = blockIdx.x, c = threadIdx.x;
  const float* xb = X + (size_t)bi * 512 * 512;
  float s = 0.f;
  for (int r = 0; r < 512; ++r) s += xb[(size_t)r * 512 + c];
  float p = s / 512.f;
  __shared__ float red[512];
  red[c] = p;
  __syncthreads();
  for (int st = 256; st > 0; st >>= 1) {
    if (c < st) red[c] += red[c + st];
    __syncthreads();
  }
  float mu = red[0] / 512.f;
  __syncthreads();
  float d = p - mu;
  red[c] = d * d;
  __syncthreads();
  for (int st = 256; st > 0; st >>= 1) {
    if (c < st) red[c] += red[c + st];
    __syncthreads();
  }
  float inv = rsqrtf(red[0] / 512.f + 1e-5f);
  float ln = (p - mu) * inv * g[c] + bt[c];
  __syncthreads();
  red[c] = ln * hw[(size_t)c * 2 + 0];
  __syncthreads();
  for (int st = 256; st > 0; st >>= 1) {
    if (c < st) red[c] += red[c + st];
    __syncthreads();
  }
  if (c == 0) out[bi * 2 + 0] = red[0] + hb[0];
  __syncthreads();
  red[c] = ln * hw[(size_t)c * 2 + 1];
  __syncthreads();
  for (int st = 256; st > 0; st >>= 1) {
    if (c < st) red[c] += red[c + st];
    __syncthreads();
  }
  if (c == 0) out[bi * 2 + 1] = red[0] + hb[1];
}

// f32 [K][N] -> f16 transposed [N][K]
__global__ __launch_bounds__(256) void cvt_h_t(const float* __restrict__ src,
                                               _Float16* __restrict__ dst,
                                               int K, int N) {
  long i = (long)blockIdx.x * 256 + threadIdx.x;
  if (i < (long)K * N) {
    int k = (int)(i / N), n = (int)(i % N);
    dst[(size_t)n * K + k] = (_Float16)src[i];
  }
}

// ---------------------------------------------------------------------------
extern "C" void kernel_launch(void* const* d_in, const int* in_sizes, int n_in,
                              void* d_out, int out_size, void* d_ws,
                              size_t ws_size, hipStream_t stream) {
  (void)in_sizes; (void)n_in; (void)out_size; (void)ws_size;
  const float* data    = (const float*)d_in[0];
  const float* latents = (const float*)d_in[1];
  const float* c_ln_g  = (const float*)d_in[2];
  const float* c_ln_b  = (const float*)d_in[3];
  const float* ctx_ln_g = (const float*)d_in[4];
  const float* ctx_ln_b = (const float*)d_in[5];
  const float* c_wq = (const float*)d_in[6];
  const float* c_wk = (const float*)d_in[7];
  const float* c_wv = (const float*)d_in[8];
  const float* c_wo = (const float*)d_in[9];
  const float* c_bo = (const float*)d_in[10];
  const float* cf_w1 = (const float*)d_in[11];
  const float* cf_b1 = (const float*)d_in[12];
  const float* cf_w2 = (const float*)d_in[13];
  const float* cf_b2 = (const float*)d_in[14];
  const float* l_ln_g = (const float*)d_in[15];
  const float* l_ln_b = (const float*)d_in[16];
  const float* l_wq = (const float*)d_in[17];
  const float* l_wk = (const float*)d_in[18];
  const float* l_wv = (const float*)d_in[19];
  const float* l_wo = (const float*)d_in[20];
  const float* l_bo = (const float*)d_in[21];
  const float* lf_w1 = (const float*)d_in[22];
  const float* lf_b1 = (const float*)d_in[23];
  const float* lf_w2 = (const float*)d_in[24];
  const float* lf_b2 = (const float*)d_in[25];
  const float* h_ln_g = (const float*)d_in[26];
  const float* h_ln_b = (const float*)d_in[27];
  const float* h_w = (const float*)d_in[28];
  const float* h_b = (const float*)d_in[29];
  float* out = (float*)d_out;

  char* ws = (char*)d_ws;
  size_t off = 0;
  auto alloc = [&](size_t bytes) -> void* {
    void* p = ws + off;
    off += (bytes + 255) & ~(size_t)255;
    return p;
  };

  const int B = 4, HW = PHW, L = 512, D = 512, FD = 2048, HD = HDI;

  _Float16* Kh  = (_Float16*)alloc((size_t)B * HW * HD * 2);
  _Float16* Vth = (_Float16*)alloc((size_t)B * HD * HW * 2);  // transposed
  _Float16* Qh  = (_Float16*)alloc((size_t)L * HD * 2);
  _Float16* Oc  = (_Float16*)alloc((size_t)B * L * HD * 2);
  _Float16* wCWO = (_Float16*)alloc((size_t)HD * D * 2);   // all weights [N][K]
  _Float16* wCF1 = (_Float16*)alloc((size_t)D * FD * 2);
  _Float16* wCF2 = (_Float16*)alloc((size_t)FD * D * 2);
  _Float16* wLQ = (_Float16*)alloc((size_t)D * D * 2);
  _Float16* wLK = (_Float16*)alloc((size_t)D * D * 2);
  _Float16* wLV = (_Float16*)alloc((size_t)D * D * 2);
  _Float16* wLO = (_Float16*)alloc((size_t)D * D * 2);
  _Float16* wLF1 = (_Float16*)alloc((size_t)D * FD * 2);
  _Float16* wLF2 = (_Float16*)alloc((size_t)FD * D * 2);
  float* x1f = (float*)alloc((size_t)B * L * D * 4);
  _Float16* x1h = (_Float16*)alloc((size_t)B * L * D * 2);
  _Float16* hch = (_Float16*)alloc((size_t)B * L * FD * 2);
  float* x2f = (float*)alloc((size_t)B * L * D * 4);
  _Float16* xnh = (_Float16*)alloc((size_t)B * L * D * 2);
  _Float16* qh = (_Float16*)alloc((size_t)B * L * D * 2);
  _Float16* kh = (_Float16*)alloc((size_t)B * L * D * 2);
  _Float16* vth = (_Float16*)alloc((size_t)B * D * L * 2);  // transposed
  _Float16* aoh = (_Float16*)alloc((size_t)B * L * D * 2);
  _Float16* x3h = (_Float16*)alloc((size_t)B * L * D * 2);
  _Float16* h2h = (_Float16*)alloc((size_t)B * L * FD * 2);
  float* x4f = (float*)alloc((size_t)B * L * D * 4);

  auto CVT = [&](const float* s, _Float16* d, int K, int N) {
    long n = (long)K * N;
    cvt_h_t<<<dim3((unsigned)((n + 255) / 256)), dim3(256), 0, stream>>>(s, d,
                                                                         K, N);
  };
  CVT(c_wo, wCWO, HD, D);
  CVT(cf_w1, wCF1, D, FD);
  CVT(cf_w2, wCF2, FD, D);
  CVT(l_wq, wLQ, D, D);
  CVT(l_wk, wLK, D, D);
  CVT(l_wv, wLV, D, D);
  CVT(l_wo, wLO, D, D);
  CVT(lf_w1, wLF1, D, FD);
  CVT(lf_w2, wLF2, FD, D);

  // Tokens -> LN -> K (row-major), V (transposed), fused
  build_kv<<<dim3((B * HW + 255) / 256), dim3(256), 0, stream>>>(
      data, ctx_ln_g, ctx_ln_b, c_wk, c_wv, Kh, Vth);
  // Q = LN(latents) @ c_wq (batch-invariant)
  q_proj<<<dim3(L), dim3(64), 0, stream>>>(latents, c_ln_g, c_ln_b, c_wq, Qh);
  // Cross attention (1 head, kv=50176); DEPTH iterations are identical -> once
  flash_attn<<<dim3(L / 64, 1, B), dim3(128), 0, stream>>>(
      Qh, Kh, Vth, Oc, HW, HD, HD, HW, HD, 0L, (long)HW * HD, (long)HD * HW,
      (long)L * HD, 0.125f);
  // x1 = O @ c_wo + c_bo
  wmma_gemm<<<dim3(D / 256, L / 16, B), dim3(128), 0, stream>>>(
      Oc, wCWO, c_bo, nullptr, x1f, x1h, L, D, HD, (long)L * HD, 0L,
      (long)L * D, 0, 0);
  // h = gelu(x1 @ cf_w1 + cf_b1)
  wmma_gemm<<<dim3(FD / 256, L / 16, B), dim3(128), 0, stream>>>(
      x1h, wCF1, cf_b1, nullptr, nullptr, hch, L, FD, D, (long)L * D, 0L,
      (long)L * FD, 1, 0);
  // x2 = h @ cf_w2 + cf_b2 + x1
  wmma_gemm<<<dim3(D / 256, L / 16, B), dim3(128), 0, stream>>>(
      hch, wCF2, cf_b2, x1f, x2f, nullptr, L, D, FD, (long)L * FD, 0L,
      (long)L * D, 0, 0);
  // xn = LN(x2)
  ln_rows<<<dim3(B * L), dim3(256), 0, stream>>>(x2f, l_ln_g, l_ln_b, xnh, D);
  // q, k row-major; v stored transposed for flash PV fragments
  wmma_gemm<<<dim3(D / 256, L / 16, B), dim3(128), 0, stream>>>(
      xnh, wLQ, nullptr, nullptr, nullptr, qh, L, D, D, (long)L * D, 0L,
      (long)L * D, 0, 0);
  wmma_gemm<<<dim3(D / 256, L / 16, B), dim3(128), 0, stream>>>(
      xnh, wLK, nullptr, nullptr, nullptr, kh, L, D, D, (long)L * D, 0L,
      (long)L * D, 0, 0);
  wmma_gemm<<<dim3(D / 256, L / 16, B), dim3(128), 0, stream>>>(
      xnh, wLV, nullptr, nullptr, nullptr, vth, L, D, D, (long)L * D, 0L,
      (long)L * D, 0, 1);
  // 8-head self attention
  flash_attn<<<dim3(L / 64, 8, B), dim3(128), 0, stream>>>(
      qh, kh, vth, aoh, L, D, D, L, D, (long)L * D, (long)L * D, (long)D * L,
      (long)L * D, 0.125f);
  // x3 = ao @ l_wo + l_bo  (no residual, faithful)
  wmma_gemm<<<dim3(D / 256, L / 16, B), dim3(128), 0, stream>>>(
      aoh, wLO, l_bo, nullptr, nullptr, x3h, L, D, D, (long)L * D, 0L,
      (long)L * D, 0, 0);
  // h2 = gelu(x3 @ lf_w1 + lf_b1)
  wmma_gemm<<<dim3(FD / 256, L / 16, B), dim3(128), 0, stream>>>(
      x3h, wLF1, lf_b1, nullptr, nullptr, h2h, L, FD, D, (long)L * D, 0L,
      (long)L * FD, 1, 0);
  // x4 = h2 @ lf_w2 + lf_b2  (no residual, faithful)
  wmma_gemm<<<dim3(D / 256, L / 16, B), dim3(128), 0, stream>>>(
      h2h, wLF2, lf_b2, nullptr, x4f, nullptr, L, D, FD, (long)L * FD, 0L,
      (long)L * D, 0, 0);
  // mean-pool + LN + linear head
  head_k<<<dim3(B), dim3(512), 0, stream>>>(x4f, h_ln_g, h_ln_b, h_w, h_b,
                                            out);
}
